// CreateUnitTypeHead_17549236371846
// MI455X (gfx1250) — matrix operand order repaired and checked by reading, not compiled, not run on hardware
//
#include <hip/hip_runtime.h>
#include <hip/hip_bf16.h>
#include <cstdint>

typedef _Float16 half_t;
typedef __attribute__((ext_vector_type(16))) _Float16 v16h;
typedef __attribute__((ext_vector_type(8)))  _Float16 v8h;
typedef __attribute__((ext_vector_type(8)))  float    v8f;
typedef __attribute__((ext_vector_type(4)))  unsigned int u32x4;
typedef __attribute__((ext_vector_type(8)))  int          i32x8;
typedef __attribute__((ext_vector_type(4)))  int          i32x4;

#define NX 512
#define NY 512
#define DD 256
#define HH 1024
#define TT 5
#define NSCALE 4
#define KCOMB 1280   // D*(1+S)
#define KCHUNK 256   // K-panel staged to LDS by the TDM
#define MTILE 64     // rows per block (4 x 16 per wave)

union ABFrag { v16h v; v8h h[2]; };

// A fragment (16x32 f16, row-major source, leading dim ld):
// lane 0-15: M=lane, halves[0..15]=K 0..15
// lane 16-31: M=lane-16, halves[0..15]=K 16..31
__device__ __forceinline__ v16h load_a_frag(const half_t* base, int ld, int lane) {
    const half_t* p = base + (size_t)(lane & 15) * ld + ((lane >> 4) << 4);
    ABFrag u;
    u.h[0] = *(const v8h*)(p);
    u.h[1] = *(const v8h*)(p + 8);
    return u.v;
}

// B fragment (32x16 f16) from a pre-TRANSPOSED weight (N x K row-major, ld=K):
// lane holds column n = lane&15; K-half = (lane>>4)*16; 16 contiguous halves.
__device__ __forceinline__ v16h load_b_frag(const half_t* baseT, int ld, int lane) {
    const half_t* p = baseT + (size_t)(lane & 15) * ld + ((lane >> 4) << 4);
    ABFrag u;
    u.h[0] = *(const v8h*)(p);
    u.h[1] = *(const v8h*)(p + 8);
    return u.v;
}

__device__ __forceinline__ v8f wmma_f16(v16h a, v16h b, v8f c) {
    return __builtin_amdgcn_wmma_f32_16x16x32_f16(
        /*neg_a=*/false, a, /*neg_b=*/false, b,
        /*c_mod=*/(short)0, c, /*reuse_a=*/false, /*reuse_b=*/false);
}

// ---------------------------------------------------------------------------
// Tensor Data Mover: 2-D tile (rows x cols f16 elements, row stride ld_halves)
// from global memory into LDS at byte offset lds_byte_off.
// D# layout per CDNA5 ISA ch.8 (group0 128b, group1 256b; groups 2/3 zero for 2D).
__device__ __forceinline__ void tdm_load_2d(uint32_t lds_byte_off, const void* gptr,
                                            int rows, int cols, int ld_halves) {
    const uint64_t ga = (uint64_t)(uintptr_t)gptr;
    u32x4 g0;
    g0[0] = 1u;                                   // count=1 (valid user descriptor)
    g0[1] = lds_byte_off;                         // lds_addr (bytes)
    g0[2] = (uint32_t)ga;                         // global_addr[31:0]
    g0[3] = (uint32_t)((ga >> 32) & 0x1FFFFFFu)   // global_addr[56:32]
            | (2u << 30);                         // type = 2 ("image")
    i32x8 g1;
    g1[0] = (1 << 16);                                            // data_size=1 (2B), mask/pad=0
    g1[1] = (cols & 0xFFFF) << 16;                                // tensor_dim0[15:0]
    g1[2] = ((cols >> 16) & 0xFFFF) | ((rows & 0xFFFF) << 16);    // tensor_dim0 hi | tensor_dim1 lo
    g1[3] = ((rows >> 16) & 0xFFFF) | ((cols & 0xFFFF) << 16);    // tensor_dim1 hi | tile_dim0
    g1[4] = (rows & 0xFFFF);                                      // tile_dim1 (tile_dim2 = 0)
    g1[5] = ld_halves;                                            // tensor_dim0_stride[31:0]
    g1[6] = 0;                                                    // stride hi / dim1_stride lo
    g1[7] = 0;
    const i32x4 gz = {0, 0, 0, 0};
#if defined(__clang_major__) && (__clang_major__ >= 23)
    const i32x8 gz8 = {0, 0, 0, 0, 0, 0, 0, 0};
    __builtin_amdgcn_tensor_load_to_lds(g0, g1, gz, gz, gz8, 0);
#else
    __builtin_amdgcn_tensor_load_to_lds(g0, g1, gz, gz, 0);
#endif
}

// ---------------------------------------------------------------------------
// f32 (K x N) -> f16 transposed (N x K)
__global__ void cvt_transpose_f16(const float* __restrict__ in,
                                  half_t* __restrict__ out, int K, int N) {
    int idx = blockIdx.x * blockDim.x + threadIdx.x;
    if (idx >= K * N) return;
    int k = idx / N;
    int n = idx - k * N;
    out[(size_t)n * K + k] = (half_t)in[idx];
}

// ---------------------------------------------------------------------------
// One block (256 thr) per city; thread = channel. Single 9x9 sweep produces
// all four mean-pools (7/5/3 patches are subsets of the 9x9 patch).
__global__ __launch_bounds__(256) void pool_kernel(
    const float* __restrict__ node_emb, const int* __restrict__ tile_ids,
    half_t* __restrict__ pooledH, half_t* __restrict__ combinedH) {
    const int city = blockIdx.x;
    const int d = threadIdx.x;             // 0..255
    const int tile = tile_ids[city];
    const int x = tile >> 9;               // / NY (512)
    const int y = tile & (NY - 1);
    float a9 = 0.f, a7 = 0.f, a5 = 0.f, a3 = 0.f;
    for (int dx = -4; dx <= 4; ++dx) {
        const int xx = x + dx;
        const bool xin = (unsigned)xx < (unsigned)NX;
        const int adx = dx < 0 ? -dx : dx;
        for (int dy = -4; dy <= 4; ++dy) {
            const int yy = y + dy;
            float v = 0.f;
            if (xin && (unsigned)yy < (unsigned)NY)
                v = node_emb[((size_t)((xx << 9) | yy)) * DD + d];
            const int ady = dy < 0 ? -dy : dy;
            const int m = adx > ady ? adx : ady;     // Chebyshev radius
            a9 += v;
            if (m <= 3) a7 += v;
            if (m <= 2) a5 += v;
            if (m <= 1) a3 += v;
        }
    }
    const size_t base = (size_t)city * (NSCALE * DD) + d;
    pooledH[base + 0 * DD] = (half_t)(a9 * (1.f / 81.f));
    pooledH[base + 1 * DD] = (half_t)(a7 * (1.f / 49.f));
    pooledH[base + 2 * DD] = (half_t)(a5 * (1.f / 25.f));
    pooledH[base + 3 * DD] = (half_t)(a3 * (1.f / 9.f));
    combinedH[(size_t)city * KCOMB + d] = (half_t)node_emb[(size_t)tile * DD + d];
}

// ---------------------------------------------------------------------------
// Fused 2-layer per-scale conv projection. Block = 64 cities x N=256.
// 8 waves, each 4 M-tiles x 2 N-tiles; layer-0 relu output staged in LDS.
__global__ __launch_bounds__(256) void conv_gemm_kernel(
    const half_t* __restrict__ pooledH,
    const half_t* __restrict__ W0T, const half_t* __restrict__ W1T,
    const float* __restrict__ b0, const float* __restrict__ b1,
    half_t* __restrict__ combinedH) {
    const int mblk = blockIdx.x;
    const int s    = blockIdx.y;
    const int wave = threadIdx.x >> 5;
    const int lane = threadIdx.x & 31;
    __shared__ half_t lds_h[MTILE * DD];      // 32 KB layer-0 output panel

    const half_t* A0 = pooledH + ((size_t)mblk * MTILE * NSCALE + s) * DD; // ld=1024
    const half_t* B0 = W0T + (size_t)s * DD * DD;
    const half_t* B1 = W1T + (size_t)s * DD * DD;
    const int n0 = wave * 32;
    const int mo = (lane >> 4) << 3;
    const int nc = lane & 15;

    // ---- layer 0: relu(pooled @ W0 + b0) -> LDS
    v8f acc[4][2] = {};
    for (int k0 = 0; k0 < DD; k0 += 32) {
        v16h bA = load_b_frag(B0 + (size_t)n0 * DD + k0, DD, lane);
        v16h bB = load_b_frag(B0 + (size_t)(n0 + 16) * DD + k0, DD, lane);
#pragma unroll
        for (int mt = 0; mt < 4; ++mt) {
            v16h a = load_a_frag(A0 + (size_t)mt * 16 * (NSCALE * DD) + k0,
                                 NSCALE * DD, lane);
            acc[mt][0] = wmma_f16(a, bA, acc[mt][0]);
            acc[mt][1] = wmma_f16(a, bB, acc[mt][1]);
        }
    }
    {
        const float g0 = b0[s * DD + n0 + nc];
        const float g1 = b0[s * DD + n0 + 16 + nc];
#pragma unroll
        for (int mt = 0; mt < 4; ++mt)
#pragma unroll
            for (int v = 0; v < 8; ++v) {
                const int r = mt * 16 + v + mo;
                lds_h[r * DD + n0 + nc]      = (half_t)fmaxf(acc[mt][0][v] + g0, 0.f);
                lds_h[r * DD + n0 + 16 + nc] = (half_t)fmaxf(acc[mt][1][v] + g1, 0.f);
            }
    }
    __syncthreads();

    // ---- layer 1: h @ W1 + b1 (no relu) -> combined[:, (1+s)*256 : ...]
    v8f dcc[4][2] = {};
    for (int k0 = 0; k0 < DD; k0 += 32) {
        v16h bA = load_b_frag(B1 + (size_t)n0 * DD + k0, DD, lane);
        v16h bB = load_b_frag(B1 + (size_t)(n0 + 16) * DD + k0, DD, lane);
#pragma unroll
        for (int mt = 0; mt < 4; ++mt) {
            v16h a = load_a_frag(lds_h + (size_t)mt * 16 * DD + k0, DD, lane);
            dcc[mt][0] = wmma_f16(a, bA, dcc[mt][0]);
            dcc[mt][1] = wmma_f16(a, bB, dcc[mt][1]);
        }
    }
    {
        const float g0 = b1[s * DD + n0 + nc];
        const float g1 = b1[s * DD + n0 + 16 + nc];
#pragma unroll
        for (int mt = 0; mt < 4; ++mt)
#pragma unroll
            for (int v = 0; v < 8; ++v) {
                const int city = mblk * MTILE + mt * 16 + v + mo;
                const size_t row = (size_t)city * KCOMB + (size_t)(1 + s) * DD;
                combinedH[row + n0 + nc]      = (half_t)(dcc[mt][0][v] + g0);
                combinedH[row + n0 + 16 + nc] = (half_t)(dcc[mt][1][v] + g1);
            }
    }
}

// ---------------------------------------------------------------------------
// MLP GEMM: out[64 x 256-tile] = act(A @ B + bias). The 64 x KCHUNK A-panel is
// DMA'd into LDS by the Tensor Data Mover (double-buffered, TENSORcnt-synced);
// all 8 waves consume it, B fragments stream from L2.
__global__ __launch_bounds__(256) void mlp_gemm_kernel(
    const half_t* __restrict__ A, int ldA, int K,
    const half_t* __restrict__ BT,            // N x K (transposed)
    const float* __restrict__ bias,
    half_t* __restrict__ out, int ldOut, int do_relu) {
    const int mblk = blockIdx.x;
    const int wave = threadIdx.x >> 5;
    const int lane = threadIdx.x & 31;
    const int n0 = blockIdx.y * 256 + wave * 32;
    __shared__ half_t ldsA[2][MTILE * KCHUNK];   // 2 x 32 KB double buffer

    const half_t* Abase = A + (size_t)mblk * MTILE * ldA;
    const int nchunks = K / KCHUNK;
    const uint32_t bufBytes = MTILE * KCHUNK * sizeof(half_t);

    if (wave == 0)
        tdm_load_2d(0u, Abase, MTILE, KCHUNK, ldA);

    v8f acc[4][2] = {};
    for (int c = 0; c < nchunks; ++c) {
        if (wave == 0) {
            if (c + 1 < nchunks) {
                tdm_load_2d((uint32_t)((c + 1) & 1) * bufBytes,
                            Abase + (size_t)(c + 1) * KCHUNK, MTILE, KCHUNK, ldA);
                __builtin_amdgcn_s_wait_tensorcnt(1);   // chunk c landed
            } else {
                __builtin_amdgcn_s_wait_tensorcnt(0);
            }
        }
        __syncthreads();                                // panel visible to all waves
        const half_t* Ac = &ldsA[c & 1][0];
        const half_t* Bc = BT + (size_t)c * KCHUNK;
        for (int k0 = 0; k0 < KCHUNK; k0 += 32) {
            v16h bA = load_b_frag(Bc + (size_t)n0 * K + k0, K, lane);
            v16h bB = load_b_frag(Bc + (size_t)(n0 + 16) * K + k0, K, lane);
#pragma unroll
            for (int mt = 0; mt < 4; ++mt) {
                v16h a = load_a_frag(Ac + (size_t)mt * 16 * KCHUNK + k0, KCHUNK, lane);
                acc[mt][0] = wmma_f16(a, bA, acc[mt][0]);
                acc[mt][1] = wmma_f16(a, bB, acc[mt][1]);
            }
        }
        __syncthreads();                                // done reading before overwrite
    }

    const int mo = (lane >> 4) << 3;
    const int nc = lane & 15;
    const float g0 = bias[n0 + nc];
    const float g1 = bias[n0 + 16 + nc];
#pragma unroll
    for (int mt = 0; mt < 4; ++mt)
#pragma unroll
        for (int v = 0; v < 8; ++v) {
            const int row = mblk * MTILE + mt * 16 + v + mo;
            float x0 = acc[mt][0][v] + g0;
            float x1 = acc[mt][1][v] + g1;
            if (do_relu) { x0 = fmaxf(x0, 0.f); x1 = fmaxf(x1, 0.f); }
            out[(size_t)row * ldOut + n0 + nc]      = (half_t)x0;
            out[(size_t)row * ldOut + n0 + 16 + nc] = (half_t)x1;
        }
}

// ---------------------------------------------------------------------------
// One wave per city: logits = h2 @ W2 + b2, masked softmax, entropy, outputs.
__global__ __launch_bounds__(256) void head_kernel(
    const half_t* __restrict__ h2, const float* __restrict__ W2,
    const float* __restrict__ b2, const int* __restrict__ create_mask,
    float* __restrict__ out, int C) {
    const int lane = threadIdx.x & 31;
    const int city = blockIdx.x * (blockDim.x >> 5) + (threadIdx.x >> 5);
    if (city >= C) return;
    const half_t* hp = h2 + (size_t)city * HH;
    float part[TT] = {0.f, 0.f, 0.f, 0.f, 0.f};
    for (int k = lane; k < HH; k += 32) {
        const float hv = (float)hp[k];
        const float* w = W2 + (size_t)k * TT;
#pragma unroll
        for (int t = 0; t < TT; ++t) part[t] += hv * w[t];
    }
#pragma unroll
    for (int t = 0; t < TT; ++t)
        for (int o = 16; o > 0; o >>= 1) part[t] += __shfl_xor(part[t], o, 32);

    if (lane == 0) {
        float logits[TT];
        int   msk[TT];
        float mx = -__builtin_inff();
#pragma unroll
        for (int t = 0; t < TT; ++t) {
            logits[t] = part[t] + b2[t];
            msk[t] = create_mask[city * TT + t] > 0;
            if (msk[t] && logits[t] > mx) mx = logits[t];
        }
        float e[TT], sum = 0.f;
#pragma unroll
        for (int t = 0; t < TT; ++t) {
            e[t] = msk[t] ? __expf(logits[t] - mx) : 0.f;
            sum += e[t];
        }
        const float inv = 1.f / sum;
        float ent = 0.f;
        const size_t oP = 0;
        const size_t oE = (size_t)C * TT;
        const size_t oL = oE + (size_t)C;
        const size_t oM = oL + (size_t)C * TT;
#pragma unroll
        for (int t = 0; t < TT; ++t) {
            const float p = e[t] * inv;
            ent -= p * __logf(fmaxf(p, 1e-8f));
            out[oP + (size_t)city * TT + t] = p;
            out[oL + (size_t)city * TT + t] = msk[t] ? logits[t] : -__builtin_inff();
            out[oM + (size_t)city * TT + t] = msk[t] ? 1.f : 0.f;
        }
        out[oE + city] = ent;
    }
}

// ---------------------------------------------------------------------------
extern "C" void kernel_launch(void* const* d_in, const int* in_sizes, int n_in,
                              void* d_out, int out_size, void* d_ws, size_t ws_size,
                              hipStream_t stream) {
    const float* node_emb = (const float*)d_in[0];
    const float* conv_W0  = (const float*)d_in[1];
    const float* conv_b0  = (const float*)d_in[2];
    const float* conv_W1  = (const float*)d_in[3];
    const float* conv_b1  = (const float*)d_in[4];
    const float* mlp_W0   = (const float*)d_in[5];
    const float* mlp_b0   = (const float*)d_in[6];
    const float* mlp_W1   = (const float*)d_in[7];
    const float* mlp_b1   = (const float*)d_in[8];
    const float* mlp_W2   = (const float*)d_in[9];
    const float* mlp_b2   = (const float*)d_in[10];
    const int*   tile_ids = (const int*)d_in[11];
    const int*   cmask    = (const int*)d_in[12];
    const int C = in_sizes[11];   // 8192

    // workspace carve-up (f16)
    half_t* pooledH   = (half_t*)d_ws;                       // C * 1024
    half_t* combinedH = pooledH   + (size_t)C * NSCALE * DD; // C * 1280
    half_t* hbuf      = combinedH + (size_t)C * KCOMB;       // C * 1024
    half_t* h2buf     = hbuf      + (size_t)C * HH;          // C * 1024
    half_t* convW0T   = h2buf     + (size_t)C * HH;          // 4*256*256
    half_t* convW1T   = convW0T   + (size_t)NSCALE * DD * DD;
    half_t* mlpW0T    = convW1T   + (size_t)NSCALE * DD * DD; // 1024 x 1280
    half_t* mlpW1T    = mlpW0T    + (size_t)HH * KCOMB;       // 1024 x 1024

    // 1) weight convert + transpose (f32 KxN -> f16 NxK)
    for (int s = 0; s < NSCALE; ++s) {
        cvt_transpose_f16<<<(DD * DD + 255) / 256, 256, 0, stream>>>(
            conv_W0 + (size_t)s * DD * DD, convW0T + (size_t)s * DD * DD, DD, DD);
        cvt_transpose_f16<<<(DD * DD + 255) / 256, 256, 0, stream>>>(
            conv_W1 + (size_t)s * DD * DD, convW1T + (size_t)s * DD * DD, DD, DD);
    }
    cvt_transpose_f16<<<(KCOMB * HH + 255) / 256, 256, 0, stream>>>(mlp_W0, mlpW0T, KCOMB, HH);
    cvt_transpose_f16<<<(HH * HH + 255) / 256, 256, 0, stream>>>(mlp_W1, mlpW1T, HH, HH);

    // 2) patch mean-pools (all scales in one sweep) + feats[0]
    pool_kernel<<<C, 256, 0, stream>>>(node_emb, tile_ids, pooledH, combinedH);

    // 3) per-scale fused 2-layer conv projections -> combined
    conv_gemm_kernel<<<dim3(C / MTILE, NSCALE), 256, 0, stream>>>(
        pooledH, convW0T, convW1T, conv_b0, conv_b1, combinedH);

    // 4) MLP layers 0 and 1 (WMMA + TDM-staged A panels)
    mlp_gemm_kernel<<<dim3(C / MTILE, HH / 256), 256, 0, stream>>>(
        combinedH, KCOMB, KCOMB, mlpW0T, mlp_b0, hbuf, HH, 1);
    mlp_gemm_kernel<<<dim3(C / MTILE, HH / 256), 256, 0, stream>>>(
        hbuf, HH, HH, mlpW1T, mlp_b1, h2buf, HH, 1);

    // 5) tiny head: logits + masked softmax + entropy + outputs
    head_kernel<<<(C + 7) / 8, 256, 0, stream>>>(h2buf, mlp_W2, mlp_b2, cmask,
                                                 (float*)d_out, C);
}
